// Decoder_13907104104759
// MI455X (gfx1250) — matrix operand order repaired and checked by reading, not compiled, not run on hardware
//
#include <hip/hip_runtime.h>
#include <math.h>

// Problem constants (match reference)
#define B_ 64
#define H_ 1024
#define E_ 1024
#define V_ 32000
#define T_ 32
#define L_ 2

typedef __bf16 bf16;
typedef __attribute__((ext_vector_type(16))) __bf16 v16bf;
typedef __attribute__((ext_vector_type(8)))  __bf16 v8bf;
typedef __attribute__((ext_vector_type(8)))  float  v8f;

// ---------------------------------------------------------------------------
// WMMA fragment loaders per CDNA5 ISA 7.12.2 layouts (wave32)
// ---------------------------------------------------------------------------

// A: 16x32 bf16 tile from row-major [.., lda] matrix, rows m0..m0+15, cols k0..k0+31.
// Lane L (0-15): M=L, elems 0..7 = K k0..k0+7, elems 8..15 = K k0+16..k0+23.
// Lane L+16:     M=L, elems 0..7 = K k0+8..k0+15, elems 8..15 = K k0+24..k0+31.
__device__ __forceinline__ v16bf load_a16x32(const bf16* __restrict__ A, int lda,
                                             int m0, int k0, int lane) {
  int row = m0 + (lane & 15);
  int kb  = k0 + ((lane & 16) ? 8 : 0);
  const v8bf* p = (const v8bf*)(A + (size_t)row * lda + kb);
  v8bf lo = p[0];   // K = kb .. kb+7
  v8bf hi = p[2];   // K = kb+16 .. kb+23
  v16bf a;
#pragma unroll
  for (int i = 0; i < 8; ++i) { a[i] = lo[i]; a[i + 8] = hi[i]; }
  return a;
}

// B = W^T tile (32x16): K rows k0..k0+31, N cols n0..n0+15, where W is
// row-major [N, ldk] (torch-style weight, so B[k][n] = W[n][k]).
// Lanes 0-15: N = n0+lane, elems 0..7 = K k0..k0+7, elems 8..15 = K k0+16..k0+23.
// Lanes 16-31: N = n0+lane-16, elems 0..7 = K k0+8..k0+15, elems 8..15 = K k0+24..k0+31.
__device__ __forceinline__ v16bf load_bT32x16(const bf16* __restrict__ W, int ldk,
                                              int n0, int k0, int lane) {
  int nrow = n0 + (lane & 15);
  int kb   = k0 + ((lane & 16) ? 8 : 0);
  const v8bf* p = (const v8bf*)(W + (size_t)nrow * ldk + kb);
  v8bf lo = p[0];   // K = kb .. kb+7
  v8bf hi = p[2];   // K = kb+16 .. kb+23
  v16bf b;
#pragma unroll
  for (int i = 0; i < 8; ++i) { b[i] = lo[i]; b[i + 8] = hi[i]; }
  return b;
}

#define WMMA_BF16(a, b, c) \
  __builtin_amdgcn_wmma_f32_16x16x32_bf16(false, (a), false, (b), (short)0, (c), false, false)

// ---------------------------------------------------------------------------
// GRU gate GEMMs: gi = x @ w_ih^T, gh = h @ w_hh^T  (M=64, N=3072, K=1024)
// grid = 24 blocks x 256 thr (8 waves); each wave owns one 16-wide N tile.
// ---------------------------------------------------------------------------
__global__ __launch_bounds__(256) void gru_gemm_kernel(
    const bf16* __restrict__ xbf,  // [64,1024]
    const bf16* __restrict__ hbf,  // [64,1024]
    const bf16* __restrict__ wih,  // [3072,1024]
    const bf16* __restrict__ whh,  // [3072,1024]
    float* __restrict__ gi,        // [64,3072]
    float* __restrict__ gh) {      // [64,3072]
  const int lane = threadIdx.x & 31;
  const int wave = threadIdx.x >> 5;
  const int n0 = (blockIdx.x * 8 + wave) * 16;

  v8f accI[4], accH[4];
#pragma unroll
  for (int mt = 0; mt < 4; ++mt) {
    accI[mt] = (v8f){0.f,0.f,0.f,0.f,0.f,0.f,0.f,0.f};
    accH[mt] = (v8f){0.f,0.f,0.f,0.f,0.f,0.f,0.f,0.f};
  }

  for (int k0 = 0; k0 < E_; k0 += 32) {
    v16bf bI = load_bT32x16(wih, E_, n0, k0, lane);
    v16bf bH = load_bT32x16(whh, H_, n0, k0, lane);
#pragma unroll
    for (int mt = 0; mt < 4; ++mt) {
      v16bf aI = load_a16x32(xbf, E_, mt * 16, k0, lane);
      accI[mt] = WMMA_BF16(aI, bI, accI[mt]);
      v16bf aH = load_a16x32(hbf, H_, mt * 16, k0, lane);
      accH[mt] = WMMA_BF16(aH, bH, accH[mt]);
    }
  }

  const int n  = n0 + (lane & 15);
  const int mh = (lane & 16) ? 8 : 0;
#pragma unroll
  for (int mt = 0; mt < 4; ++mt) {
    const int mb = mt * 16 + mh;
#pragma unroll
    for (int r = 0; r < 8; ++r) {
      gi[(size_t)(mb + r) * (3 * H_) + n] = accI[mt][r];
      gh[(size_t)(mb + r) * (3 * H_) + n] = accH[mt][r];
    }
  }
}

// ---------------------------------------------------------------------------
// Logits GEMM: logits = x @ out_w^T + out_b  (M=64, N=32000, K=1024)
// grid = 250 blocks x 256 thr (8 waves). Writes directly into d_out[b, t, n].
// ---------------------------------------------------------------------------
__global__ __launch_bounds__(256) void logits_gemm_kernel(
    const bf16* __restrict__ x,     // [64,1024] (h of layer 1)
    const bf16* __restrict__ outw,  // [32000,1024]
    const float* __restrict__ outb, // [32000]
    float* __restrict__ out,        // [B, T, V] fp32
    int t) {
  const int lane = threadIdx.x & 31;
  const int wave = threadIdx.x >> 5;
  const int n0 = (blockIdx.x * 8 + wave) * 16;

  v8f acc[4];
#pragma unroll
  for (int mt = 0; mt < 4; ++mt)
    acc[mt] = (v8f){0.f,0.f,0.f,0.f,0.f,0.f,0.f,0.f};

  const bf16* wrow = outw + (size_t)(n0 + (lane & 15)) * H_;
  for (int k0 = 0; k0 < H_; k0 += 32) {
    if (k0 + 64 < H_) __builtin_prefetch(wrow + k0 + 64, 0, 0); // K tile two steps ahead
    v16bf bb = load_bT32x16(outw, H_, n0, k0, lane);
#pragma unroll
    for (int mt = 0; mt < 4; ++mt) {
      v16bf aa = load_a16x32(x, H_, mt * 16, k0, lane);
      acc[mt] = WMMA_BF16(aa, bb, acc[mt]);
    }
  }

  const int n  = n0 + (lane & 15);
  const float bias = outb[n];
  const int mh = (lane & 16) ? 8 : 0;
  const size_t tcol = (size_t)t * V_ + n;
#pragma unroll
  for (int mt = 0; mt < 4; ++mt) {
    const int mb = mt * 16 + mh;
#pragma unroll
    for (int r = 0; r < 8; ++r)
      out[(size_t)(mb + r) * ((size_t)T_ * V_) + tcol] = acc[mt][r] + bias;
  }
}

// ---------------------------------------------------------------------------
// Elementwise GRU gate math (fp32, matches reference; r,z,n gate order)
// ---------------------------------------------------------------------------
__global__ __launch_bounds__(256) void gru_gates_kernel(
    const float* __restrict__ gi, const float* __restrict__ gh,
    const float* __restrict__ bi, const float* __restrict__ bh,
    float* __restrict__ h, bf16* __restrict__ hbf) {
  const int i = blockIdx.x * 256 + threadIdx.x;  // 0 .. 64*1024-1
  const int b = i >> 10, j = i & 1023;
  const size_t g = (size_t)b * (3 * H_) + j;
  const float ir  = gi[g]            + bi[j];
  const float iz  = gi[g + H_]       + bi[j + H_];
  const float inn = gi[g + 2 * H_]   + bi[j + 2 * H_];
  const float hr  = gh[g]            + bh[j];
  const float hz  = gh[g + H_]       + bh[j + H_];
  const float hn  = gh[g + 2 * H_]   + bh[j + 2 * H_];
  const float r = 1.f / (1.f + expf(-(ir + hr)));
  const float z = 1.f / (1.f + expf(-(iz + hz)));
  const float n = tanhf(inn + r * hn);
  const float hv = (1.f - z) * n + z * h[i];
  h[i] = hv;
  hbf[i] = (bf16)hv;
}

// ---------------------------------------------------------------------------
// Embedding gather + ReLU -> bf16 activations
// ---------------------------------------------------------------------------
__global__ __launch_bounds__(256) void embed_relu_kernel(
    const float* __restrict__ emb, const int* __restrict__ tok,
    bf16* __restrict__ x) {
  const int i = blockIdx.x * 256 + threadIdx.x;  // 64*1024
  const int b = i >> 10, e = i & 1023;
  const float v = emb[(size_t)tok[b] * E_ + e];
  x[i] = (bf16)(v > 0.f ? v : 0.f);
}

// ---------------------------------------------------------------------------
// Per-batch argmax over V logits (first-index tie-break like jnp.argmax)
// ---------------------------------------------------------------------------
__global__ __launch_bounds__(256) void argmax_kernel(
    const float* __restrict__ out, int t, int* __restrict__ tok) {
  const int b = blockIdx.x;
  const float* row = out + (size_t)b * ((size_t)T_ * V_) + (size_t)t * V_;
  __shared__ float sval[256];
  __shared__ int   sidx[256];
  float best = -3.402823466e38f;
  int   bidx = 0x7fffffff;
  for (int n = threadIdx.x; n < V_; n += 256) {
    float v = row[n];
    if (v > best || (v == best && n < bidx)) { best = v; bidx = n; }
  }
  sval[threadIdx.x] = best; sidx[threadIdx.x] = bidx;
  __syncthreads();
  for (int s = 128; s > 0; s >>= 1) {
    if (threadIdx.x < s) {
      float v = sval[threadIdx.x + s]; int i2 = sidx[threadIdx.x + s];
      if (v > sval[threadIdx.x] ||
          (v == sval[threadIdx.x] && i2 < sidx[threadIdx.x])) {
        sval[threadIdx.x] = v; sidx[threadIdx.x] = i2;
      }
    }
    __syncthreads();
  }
  if (threadIdx.x == 0) tok[b] = sidx[0];
}

// ---------------------------------------------------------------------------
// Setup / teardown helpers
// ---------------------------------------------------------------------------
__global__ __launch_bounds__(256) void f32_to_bf16_kernel(
    const float* __restrict__ in, bf16* __restrict__ out, int n) {
  const int i = blockIdx.x * 256 + threadIdx.x;
  if (i < n) out[i] = (bf16)in[i];
}

__global__ __launch_bounds__(256) void init_h_kernel(
    const float* __restrict__ src, float* __restrict__ h,
    bf16* __restrict__ hbf, int n) {
  const int i = blockIdx.x * 256 + threadIdx.x;
  if (i < n) { float v = src[i]; h[i] = v; hbf[i] = (bf16)v; }
}

__global__ void init_tok_kernel(int* __restrict__ tok) {
  if (threadIdx.x < B_) tok[threadIdx.x] = 0;  // START_TOKEN
}

__global__ __launch_bounds__(256) void copy_h_out_kernel(
    const float* __restrict__ h, float* __restrict__ out, int n) {
  const int i = blockIdx.x * 256 + threadIdx.x;
  if (i < n) out[i] = h[i];
}

// ---------------------------------------------------------------------------
// Host orchestration
// ---------------------------------------------------------------------------
extern "C" void kernel_launch(void* const* d_in, const int* in_sizes, int n_in,
                              void* d_out, int out_size, void* d_ws, size_t ws_size,
                              hipStream_t stream) {
  (void)in_sizes; (void)n_in; (void)out_size; (void)ws_size;
  // setup_inputs order:
  // 0 encoder_outputs (unused by reference), 1 encoder_hidden, 2 emb,
  // 3 w_ih, 4 w_hh, 5 b_ih, 6 b_hh, 7 out_w, 8 out_b, 9 max_length
  const float* enc_hid = (const float*)d_in[1];
  const float* emb     = (const float*)d_in[2];
  const float* w_ih    = (const float*)d_in[3];
  const float* w_hh    = (const float*)d_in[4];
  const float* b_ih    = (const float*)d_in[5];
  const float* b_hh    = (const float*)d_in[6];
  const float* out_w   = (const float*)d_in[7];
  const float* out_b   = (const float*)d_in[8];
  float* out = (float*)d_out;

  // Workspace carve-out (~93 MB; fits in 192 MB L2 alongside streaming output)
  char* ws = (char*)d_ws;
  size_t off = 0;
  auto take = [&](size_t bytes) -> void* {
    off = (off + 255) & ~(size_t)255;
    void* p = ws + off;
    off += bytes;
    return p;
  };
  const int WIH_N  = L_ * 3 * H_ * E_;   // 6,291,456
  const int WHH_N  = L_ * 3 * H_ * H_;   // 6,291,456
  const int OUTW_N = V_ * H_;            // 32,768,000
  const int HN     = L_ * B_ * H_;       // 131,072
  bf16*  wih_bf  = (bf16*) take((size_t)WIH_N * 2);
  bf16*  whh_bf  = (bf16*) take((size_t)WHH_N * 2);
  bf16*  outw_bf = (bf16*) take((size_t)OUTW_N * 2);
  float* hbuf    = (float*)take((size_t)HN * 4);
  bf16*  hbf     = (bf16*) take((size_t)HN * 2);
  bf16*  xbf     = (bf16*) take((size_t)B_ * E_ * 2);
  float* gi      = (float*)take((size_t)B_ * 3 * H_ * 4);
  float* gh      = (float*)take((size_t)B_ * 3 * H_ * 4);
  int*   tok     = (int*)  take(B_ * 4);

  // One-time (per launch) weight down-conversion to bf16
  f32_to_bf16_kernel<<<(WIH_N + 255) / 256, 256, 0, stream>>>(w_ih, wih_bf, WIH_N);
  f32_to_bf16_kernel<<<(WHH_N + 255) / 256, 256, 0, stream>>>(w_hh, whh_bf, WHH_N);
  f32_to_bf16_kernel<<<(OUTW_N + 255) / 256, 256, 0, stream>>>(out_w, outw_bf, OUTW_N);
  init_h_kernel<<<(HN + 255) / 256, 256, 0, stream>>>(enc_hid, hbuf, hbf, HN);
  init_tok_kernel<<<1, 64, 0, stream>>>(tok);

  const int BH = B_ * H_;  // 65,536 elems per [B,H] slab
  for (int t = 0; t < T_; ++t) {
    embed_relu_kernel<<<BH / 256, 256, 0, stream>>>(emb, tok, xbf);

    // Layer 0: A_x = relu(emb), A_h = h[0]
    gru_gemm_kernel<<<24, 256, 0, stream>>>(xbf, hbf,
                                            wih_bf, whh_bf, gi, gh);
    gru_gates_kernel<<<BH / 256, 256, 0, stream>>>(gi, gh, b_ih, b_hh,
                                                   hbuf, hbf);
    // Layer 1: A_x = h[0] (fresh), A_h = h[1]
    gru_gemm_kernel<<<24, 256, 0, stream>>>(hbf, hbf + BH,
                                            wih_bf + (size_t)3 * H_ * E_,
                                            whh_bf + (size_t)3 * H_ * H_,
                                            gi, gh);
    gru_gates_kernel<<<BH / 256, 256, 0, stream>>>(gi, gh,
                                                   b_ih + 3 * H_, b_hh + 3 * H_,
                                                   hbuf + BH, hbf + BH);

    // logits -> d_out[:, t, :] (+bias), then greedy token
    logits_gemm_kernel<<<250, 256, 0, stream>>>(hbf + BH, outw_bf, out_b, out, t);
    argmax_kernel<<<B_, 256, 0, stream>>>(out, t, tok);
  }

  // h_final [L,B,H] appended after decoder_outputs [B,T,V]
  copy_h_out_kernel<<<(HN + 255) / 256, 256, 0, stream>>>(
      hbuf, out + (size_t)B_ * T_ * V_, HN);
}
